// ControllerDecoder_34626026341014
// MI455X (gfx1250) — compile-verified, matrix operand order checked by reading, not run on hardware
//
#include <hip/hip_runtime.h>

// ---------------------------------------------------------------------------
// ControllerDecoder.sample fused kernel for MI455X (gfx1250, wave32, WMMA).
//
// Heads: n = {21,21,21,21,5,32}, K_i = 512 + 64*i.
// logits_i = x @ Wi[0:512] + sum_{j<i} emb_j @ Wi[512+64j : 512+64(j+1)] + b_i
// GEMMs run as v_wmma_f32_16x16x32_bf16 with N padded to 32 (2 N-tiles).
// x is pre-converted to bf16 in workspace (L2-resident, 128 MB).
// Each wave32 owns 32 rows (2 M-tiles) end-to-end so every B-fragment pair
// feeds 4 WMMAs: 8 b128 loads -> 4 WMMAs in the inner loop.
// No inter-wave synchronization anywhere.
// ---------------------------------------------------------------------------

typedef __attribute__((ext_vector_type(16))) __bf16 v16bf;
typedef __attribute__((ext_vector_type(8)))  float  v8f;

union Frag {
    unsigned u[8];   // 8 dwords == 16 packed bf16
    v16bf    v;
};

__device__ __forceinline__ unsigned bf16_rne(float f) {
    unsigned u = __float_as_uint(f);
    return ((u + 0x7FFFu + ((u >> 16) & 1u)) >> 16) & 0xFFFFu;
}
__device__ __forceinline__ unsigned pack_bf16x2(float lo, float hi) {
    return bf16_rne(lo) | (bf16_rne(hi) << 16);
}
__device__ __forceinline__ unsigned pcg_hash(unsigned v) {
    v = v * 747796405u + 2891336453u;
    v = ((v >> ((v >> 28) + 4u)) ^ v) * 277803737u;
    return (v >> 22) ^ v;
}

#define NEG_INF (-__builtin_inff())

// --- prep: weights [K][n] f32 -> column-major padded [32][K] bf16 (dwords) ---
__global__ void cvt_weights_kernel(const float* __restrict__ W,
                                   unsigned* __restrict__ dst, int K, int n) {
    int t = blockIdx.x * 256 + threadIdx.x;
    int kd = K >> 1;                       // dwords per column
    if (t >= 32 * kd) return;
    int col = t / kd;
    int k2  = t - col * kd;
    float f0 = 0.f, f1 = 0.f;
    if (col < n) {
        f0 = W[(2 * k2) * n + col];
        f1 = W[(2 * k2 + 1) * n + col];
    }
    dst[t] = pack_bf16x2(f0, f1);
}

// --- prep: embed [89][64] f32 -> bf16 packed dwords [89*32] ---
__global__ void cvt_embed_kernel(const float* __restrict__ e,
                                 unsigned* __restrict__ dst, int ndw) {
    int t = blockIdx.x * 256 + threadIdx.x;
    if (t < ndw) dst[t] = pack_bf16x2(e[2 * t], e[2 * t + 1]);
}

// --- prep: x [B][512] f32 -> bf16 packed dwords [B][256] (streaming) ---
__global__ void cvt_x_kernel(const float* __restrict__ x,
                             unsigned* __restrict__ dst, long long ndw4) {
    long long t = (long long)blockIdx.x * 256 + threadIdx.x;   // uint4 index
    if (t >= ndw4) return;
    const float4 f0 = *(const float4*)(x + t * 8);
    const float4 f1 = *(const float4*)(x + t * 8 + 4);
    uint4 o;
    o.x = pack_bf16x2(f0.x, f0.y);
    o.y = pack_bf16x2(f0.z, f0.w);
    o.z = pack_bf16x2(f1.x, f1.y);
    o.w = pack_bf16x2(f1.z, f1.w);
    *(uint4*)(dst + t * 4) = o;
}

// B-fragment load: wtHead = column-major [32][K] bf16 as dwords.
__device__ __forceinline__ void load_bfrag(Frag& f, const unsigned* __restrict__ wtHead,
                                           int K, int col, int k0, int sel) {
    const unsigned* base = wtHead + col * (K >> 1) + (k0 >> 1) + sel * 4;
    uint4 c0 = *(const uint4*)(base);
    uint4 c1 = *(const uint4*)(base + 8);
    f.u[0] = c0.x; f.u[1] = c0.y; f.u[2] = c0.z; f.u[3] = c0.w;
    f.u[4] = c1.x; f.u[5] = c1.y; f.u[6] = c1.z; f.u[7] = c1.w;
}

// A-fragment from packed bf16 dwords (row base pointer, 2 chunks per lane).
__device__ __forceinline__ void load_afrag_bf(Frag& f, const unsigned* __restrict__ rowp,
                                              int k0, int sel) {
    uint4 c0 = *(const uint4*)(rowp + (k0 >> 1) + sel * 4);
    uint4 c1 = *(const uint4*)(rowp + (k0 >> 1) + 8 + sel * 4);
    f.u[0] = c0.x; f.u[1] = c0.y; f.u[2] = c0.z; f.u[3] = c0.w;
    f.u[4] = c1.x; f.u[5] = c1.y; f.u[6] = c1.z; f.u[7] = c1.w;
}

// A-fragment from f32 row with on-the-fly bf16 conversion (fallback path).
__device__ __forceinline__ void load_afrag_f32(Frag& a, const float* __restrict__ xrow,
                                               int k0, int sel) {
    const float4 f0 = *(const float4*)(xrow + k0 + sel * 8);
    const float4 f1 = *(const float4*)(xrow + k0 + sel * 8 + 4);
    const float4 f2 = *(const float4*)(xrow + k0 + 16 + sel * 8);
    const float4 f3 = *(const float4*)(xrow + k0 + 16 + sel * 8 + 4);
    a.u[0] = pack_bf16x2(f0.x, f0.y); a.u[1] = pack_bf16x2(f0.z, f0.w);
    a.u[2] = pack_bf16x2(f1.x, f1.y); a.u[3] = pack_bf16x2(f1.z, f1.w);
    a.u[4] = pack_bf16x2(f2.x, f2.y); a.u[5] = pack_bf16x2(f2.z, f2.w);
    a.u[6] = pack_bf16x2(f3.x, f3.y); a.u[7] = pack_bf16x2(f3.z, f3.w);
}

#define WMMA_BF16(A, Bf, C) \
    __builtin_amdgcn_wmma_f32_16x16x32_bf16(false, (A).v, false, (Bf).v, (short)0, (C), false, false)

template <bool XBF16>
__global__ __launch_bounds__(64)
void decoder_kernel(const float*    __restrict__ x,       // [B][512] f32 (fallback)
                    const unsigned* __restrict__ xbf,     // [B][256] bf16 dwords
                    const unsigned* __restrict__ embw,    // bf16 embed dwords [89*32]
                    const unsigned* __restrict__ wt,      // all heads col-major bf16
                    const float* __restrict__ b0, const float* __restrict__ b1,
                    const float* __restrict__ b2, const float* __restrict__ b3,
                    const float* __restrict__ b4, const float* __restrict__ b5,
                    float* __restrict__ out, int B) {
    constexpr int NS[6]  = {21, 21, 21, 21, 5, 32};
    constexpr int OFF[5] = {0, 21, 42, 63, 84};
    constexpr int WOFF[6] = {0, 8192, 17408, 27648, 38912, 51200}; // dword offsets

    __shared__ unsigned featsLds[64 * 5 * 32]; // [row][slot j][32 dwords = 64 bf16]
    __shared__ float    logitsLds[64 * 32];
    __shared__ float    entLds[64];

    const int lane    = threadIdx.x & 31;
    const int wave    = threadIdx.x >> 5;          // 0..1
    const int rowBase = wave * 32;                 // wave owns 32 rows (2 M-tiles)
    const int sel     = lane >> 4;                 // K-half select per WMMA layout
    const int col0    = lane & 15;
    const long long tile0 = (long long)blockIdx.x * 64;
    const long long grow0 = tile0 + rowBase + col0;        // M-tile 0 A row
    const long long grow1 = grow0 + 16;                    // M-tile 1 A row

    const float* biases[6] = {b0, b1, b2, b3, b4, b5};

    #pragma unroll 1
    for (int i = 0; i < 6; ++i) {
        const int K = 512 + 64 * i;
        const unsigned* wtHead = wt + WOFF[i];
        __builtin_prefetch(wtHead, 0, 0);          // global_prefetch_b8

        v8f acc00 = {}, acc01 = {};   // rows 0-15  x cols 0-15 / 16-31
        v8f acc10 = {}, acc11 = {};   // rows 16-31 x cols 0-15 / 16-31

        // ---- x contribution: K = 0..511 ----
        const float*    xr0  = x   + grow0 * 512;
        const float*    xr1  = x   + grow1 * 512;
        const unsigned* xb0  = xbf + grow0 * 256;
        const unsigned* xb1  = xbf + grow1 * 256;
        #pragma unroll 4
        for (int k0 = 0; k0 < 512; k0 += 32) {
            Frag a0, a1, bb0, bb1;
            if (XBF16) {
                load_afrag_bf(a0, xb0, k0, sel);
                load_afrag_bf(a1, xb1, k0, sel);
            } else {
                load_afrag_f32(a0, xr0, k0, sel);
                load_afrag_f32(a1, xr1, k0, sel);
            }
            load_bfrag(bb0, wtHead, K, col0,      k0, sel);
            load_bfrag(bb1, wtHead, K, col0 + 16, k0, sel);
            acc00 = WMMA_BF16(a0, bb0, acc00);
            acc01 = WMMA_BF16(a0, bb1, acc01);
            acc10 = WMMA_BF16(a1, bb0, acc10);
            acc11 = WMMA_BF16(a1, bb1, acc11);
        }

        // ---- autoregressive embedding contributions (A from LDS feats) ----
        for (int j = 0; j < i; ++j) {
            const unsigned* fr0 = featsLds + (rowBase + col0) * 160 + j * 32;
            const unsigned* fr1 = fr0 + 16 * 160;
            #pragma unroll
            for (int k0 = 0; k0 < 64; k0 += 32) {
                Frag a0, a1, bb0, bb1;
                load_afrag_bf(a0, fr0, k0, sel);
                load_afrag_bf(a1, fr1, k0, sel);
                load_bfrag(bb0, wtHead, K, col0,      512 + j * 64 + k0, sel);
                load_bfrag(bb1, wtHead, K, col0 + 16, 512 + j * 64 + k0, sel);
                acc00 = WMMA_BF16(a0, bb0, acc00);
                acc01 = WMMA_BF16(a0, bb1, acc01);
                acc10 = WMMA_BF16(a1, bb0, acc10);
                acc11 = WMMA_BF16(a1, bb1, acc11);
            }
        }

        // ---- transpose accumulators into LDS (C layout: VGPR r -> M=r / M=8+r) ----
        #pragma unroll
        for (int r = 0; r < 8; ++r) {
            int row = rowBase + r + sel * 8;
            logitsLds[row * 32 + col0]             = acc00[r];
            logitsLds[row * 32 + col0 + 16]        = acc01[r];
            logitsLds[(row + 16) * 32 + col0]      = acc10[r];
            logitsLds[(row + 16) * 32 + col0 + 16] = acc11[r];
        }

        // ---- per-row softmax entropy, top-5, Gumbel sample, gather (lane = col) ----
        const int   n     = NS[i];
        const bool  valid = lane < n;
        const float bias  = valid ? biases[i][lane] : 0.f;

        for (int r = 0; r < 32; ++r) {
            const int row = rowBase + r;
            const long long growr = tile0 + row;
            float l = valid ? (logitsLds[row * 32 + lane] + bias) : NEG_INF;

            float mx = l;
            #pragma unroll
            for (int d = 16; d; d >>= 1) mx = fmaxf(mx, __shfl_xor(mx, d, 32));
            float e  = valid ? __expf(l - mx) : 0.f;
            float tt = valid ? e * (l - mx)   : 0.f;
            float s1 = e, s2 = tt;
            #pragma unroll
            for (int d = 16; d; d >>= 1) {
                s1 += __shfl_xor(s1, d, 32);
                s2 += __shfl_xor(s2, d, 32);
            }
            const float H = __logf(s1) - s2 / s1;   // categorical entropy

            // top-5 threshold: 5 successive argmax removals
            float cur = l, kth = NEG_INF;
            #pragma unroll
            for (int it = 0; it < 5; ++it) {
                float m = cur;
                #pragma unroll
                for (int d = 16; d; d >>= 1) m = fmaxf(m, __shfl_xor(m, d, 32));
                kth = m;
                unsigned long long bal = __ballot(cur == m);
                int slane = bal ? (int)__builtin_ctzll(bal) : 0;
                if (lane == slane) cur = NEG_INF;
            }

            // Gumbel-max sampling over top-5-masked logits
            unsigned h = pcg_hash(((unsigned)growr * 6u + (unsigned)i) * 37u + lane + 42u);
            float u = (float)(h >> 8) * (1.f / 16777216.f) + 1e-10f;
            float g = -__logf(-__logf(u));
            float score = (valid && l >= kth) ? (l + g) : NEG_INF;
            float sm = score;
            #pragma unroll
            for (int d = 16; d; d >>= 1) sm = fmaxf(sm, __shfl_xor(sm, d, 32));
            unsigned long long bal2 = __ballot(score == sm);
            int action = bal2 ? (int)__builtin_ctzll(bal2) : 0;

            if (lane == 0) {
                out[growr * 6 + i] = (float)action;
                float eAcc = (i == 0 ? 0.f : entLds[row]) + H;
                if (i == 5) out[(long long)B * 6 + growr] = eAcc;
                else        entLds[row] = eAcc;
            }

            // gather bf16 embedding row into feats slot i (32 dwords = 64 bf16)
            if (i < 5) {
                int erow = action + OFF[i];
                featsLds[row * 160 + i * 32 + lane] = embw[erow * 32 + lane];
            }
        }
    }
}

extern "C" void kernel_launch(void* const* d_in, const int* in_sizes, int n_in,
                              void* d_out, int out_size, void* d_ws, size_t ws_size,
                              hipStream_t stream) {
    const float* x     = (const float*)d_in[0];
    const float* embed = (const float*)d_in[1];
    const float* Ws[6] = {(const float*)d_in[2], (const float*)d_in[3],
                          (const float*)d_in[4], (const float*)d_in[5],
                          (const float*)d_in[6], (const float*)d_in[7]};
    const float* bs[6] = {(const float*)d_in[8],  (const float*)d_in[9],
                          (const float*)d_in[10], (const float*)d_in[11],
                          (const float*)d_in[12], (const float*)d_in[13]};
    const int B = in_sizes[0] / 512;

    static const int Ks[6] = {512, 576, 640, 704, 768, 832};
    static const int Ns[6] = {21, 21, 21, 21, 5, 32};

    // workspace layout (dwords):
    //   wt   : 64512  (all heads, col-major bf16, padded to 32 cols)
    //   embw : 2848   (89x64 bf16)
    //   xbf  : B*256  (x in bf16) -- only if ws_size is large enough
    unsigned* wt   = (unsigned*)d_ws;
    unsigned* embw = wt + 64512;
    unsigned* xbf  = embw + 2848;

    const size_t needBase = (size_t)(64512 + 2848) * 4;
    const size_t needFull = needBase + (size_t)B * 256 * 4;
    const bool useXbf16 = (ws_size >= needFull);

    int woff = 0;
    for (int i = 0; i < 6; ++i) {
        int ndw = 32 * (Ks[i] >> 1);
        cvt_weights_kernel<<<(ndw + 255) / 256, 256, 0, stream>>>(Ws[i], wt + woff,
                                                                  Ks[i], Ns[i]);
        woff += ndw;
    }
    cvt_embed_kernel<<<(2848 + 255) / 256, 256, 0, stream>>>(embed, embw, 2848);

    if (useXbf16) {
        long long ndw4 = (long long)B * 64;   // uint4 chunks of x_bf16
        cvt_x_kernel<<<(unsigned)((ndw4 + 255) / 256), 256, 0, stream>>>(x, xbf, ndw4);
        decoder_kernel<true><<<B / 64, 64, 0, stream>>>(x, xbf, embw, wt,
                                                        bs[0], bs[1], bs[2],
                                                        bs[3], bs[4], bs[5],
                                                        (float*)d_out, B);
    } else {
        decoder_kernel<false><<<B / 64, 64, 0, stream>>>(x, xbf, embw, wt,
                                                         bs[0], bs[1], bs[2],
                                                         bs[3], bs[4], bs[5],
                                                         (float*)d_out, B);
    }
}
